// VectorQuantizer_18717467476655
// MI455X (gfx1250) — matrix operand order repaired and checked by reading, not compiled, not run on hardware
//
#include <hip/hip_runtime.h>
#include <hip/hip_bf16.h>
#include <math.h>

// ---- problem constants ----
#define TOK   16384      // 8*2048 tokens
#define KC    8192       // codebook entries
#define DIM   256        // embedding dim
#define TD    (TOK*DIM)  // 4194304
#define KD    (KC*DIM)   // 2097152

#define KSTEP 32         // codebook rows per TDM tile
#define NITER (KC/KSTEP) // 256 k-iterations
#define TILEB (KSTEP*DIM*2)  // 16384 bytes per B tile

typedef __attribute__((ext_vector_type(16))) _Float16 v16h;
typedef __attribute__((ext_vector_type(8)))  _Float16 v8h;
typedef __attribute__((ext_vector_type(8)))  float    v8f;
typedef __attribute__((ext_vector_type(4)))  unsigned int u32x4;
typedef __attribute__((ext_vector_type(8)))  int          i32x8;
typedef __attribute__((ext_vector_type(4)))  int          i32x4;

#ifndef __has_builtin
#define __has_builtin(x) 0
#endif
#if __has_builtin(__builtin_amdgcn_tensor_load_to_lds)
#define HAVE_TDM 1
#else
#define HAVE_TDM 0
#endif

// -------------------------------------------------------------------------
// Kernel 1: prep — f32->f16 conversion of X and E, ||e||^2, zero accumulators
// -------------------------------------------------------------------------
__global__ __launch_bounds__(256) void vq_prep(const float* __restrict__ x,
                                               const float* __restrict__ e,
                                               _Float16* __restrict__ xh,
                                               _Float16* __restrict__ eh,
                                               float* __restrict__ esq,
                                               float* __restrict__ avg,
                                               float* __restrict__ accSq) {
    int i = blockIdx.x * 256 + threadIdx.x;
    xh[i] = (_Float16)x[i];
    if (i < KD) eh[i] = (_Float16)e[i];
    if (i < KC) {
        const float* er = e + i * DIM;
        float s = 0.f;
        #pragma unroll 4
        for (int d = 0; d < DIM; ++d) { float v = er[d]; s = fmaf(v, v, s); }
        esq[i] = s;
        avg[i] = 0.f;
    }
    if (i == 0) accSq[0] = 0.f;
}

#if HAVE_TDM
// Issue a TDM DMA of one B tile: KSTEP rows x 256 f16 (row-major, contiguous)
// from global eh into LDS at byte offset lds_off.  D# per CDNA5 ISA section 8.
static __device__ __forceinline__ void tdm_load_tile(unsigned long long ga, unsigned lds_off) {
    u32x4 g0 = { 1u,                                   // count=1, user descriptor
                 lds_off,                              // lds_addr (bytes)
                 (unsigned)ga,                         // global_addr[31:0]
                 ((unsigned)(ga >> 32) & 0x01FFFFFFu)  // global_addr[56:32]
                 | 0x80000000u };                      // type=2 ("image")
    i32x8 g1 = { (int)0x00010000,        // wg_mask=0, data_size=1 (2 bytes)
                 (int)(256u << 16),      // tensor_dim0[15:0]=256 in [31:16]
                 (int)(8192u << 16),     // tensor_dim0 hi=0 | tensor_dim1 lo=8192
                 (int)(256u << 16),      // tensor_dim1 hi=0 | tile_dim0=256
                 (int)KSTEP,             // tile_dim1=KSTEP, tile_dim2=0
                 (int)256,               // tensor_dim0_stride lo = 256 elements
                 0, 0 };                 // stride hi / dim1_stride = 0
    i32x4 gz = { 0, 0, 0, 0 };
#if __clang_major__ >= 23
    i32x8 gz8 = { 0, 0, 0, 0, 0, 0, 0, 0 };
    __builtin_amdgcn_tensor_load_to_lds(g0, g1, gz, gz, gz8, 0);
#else
    __builtin_amdgcn_tensor_load_to_lds(g0, g1, gz, gz, 0);
#endif
}
#endif

// -------------------------------------------------------------------------
// GEMM passes. Block = 256 threads (8 waves), owns 128 tokens x all K.
// wave w: wm = w&3 picks a 32-token group; wk = w>>2 picks a 16-row k-subtile.
// Each wave keeps TWO 16-token A tiles resident in VGPRs (128 regs); B tiles
// (KSTEP rows x 256 f16) are DMA'd by the Tensor Data Mover into LDS, double
// buffered, and consumed via ds_load by all 8 waves (4x reuse of each tile).
// PASS==1: flash-softmax (running max, online Z, argmax) -> gM,gZ,gI
// PASS==2: p = exp(s - M)/Z accumulated into LDS[8192] then global atomics
// dynamic LDS: [0,16K) bufA | [16K,32K) bufB | PASS2: [32K,64K) lacc
// -------------------------------------------------------------------------
template <int PASS>
__global__ __launch_bounds__(256) void vq_gemm(const _Float16* __restrict__ xh,
                                               const _Float16* __restrict__ eh,
                                               const float* __restrict__ esq,
                                               float* __restrict__ gM,
                                               float* __restrict__ gZ,
                                               int*   __restrict__ gI,
                                               float* __restrict__ avg) {
    extern __shared__ char smem[];
    float* lacc = (float*)(smem + 2 * TILEB);
    __shared__ float sM[128][2];
    __shared__ float sZ[128][2];
    __shared__ int   sB[128][2];

    const int tid  = threadIdx.x;
    const int w    = tid >> 5;
    const int lane = tid & 31;
    const int ln   = lane & 15;
    const int hf   = lane >> 4;
    const int wm   = w & 3;                    // 0..3 token group
    const int wk   = w >> 2;                   // 0..1 k-subtile
    const int tb   = blockIdx.x * 128 + wm * 32;  // this wave's 32 tokens

    // ---- A fragments: 2 tiles x 16 tokens x 256 dims, resident in regs.
    // ISA 16-bit A layout: lanes 0-15 hold K {0..7,16..23}, lanes 16-31 {8..15,24..31}
    v16h af0[8], af1[8];
    {
        const _Float16* xr0 = xh + (size_t)(tb + ln) * DIM + 8 * hf;
        const _Float16* xr1 = xh + (size_t)(tb + 16 + ln) * DIM + 8 * hf;
        #pragma unroll
        for (int c = 0; c < 8; ++c) {
            v8h lo = *(const v8h*)(xr0 + 32 * c);
            v8h hi = *(const v8h*)(xr0 + 32 * c + 16);
            af0[c] = __builtin_shufflevector(lo, hi, 0,1,2,3,4,5,6,7,8,9,10,11,12,13,14,15);
            lo = *(const v8h*)(xr1 + 32 * c);
            hi = *(const v8h*)(xr1 + 32 * c + 16);
            af1[c] = __builtin_shufflevector(lo, hi, 0,1,2,3,4,5,6,7,8,9,10,11,12,13,14,15);
        }
    }

    float mx[2][8], zs[2][8], Mv[2][8], iZ[2][8];
    int   bi[2][8];
    #pragma unroll
    for (int u = 0; u < 2; ++u)
        #pragma unroll
        for (int r = 0; r < 8; ++r) { mx[u][r] = -INFINITY; zs[u][r] = 0.f; bi[u][r] = 0; }

    if (PASS == 2) {
        for (int i = tid; i < KC; i += 256) lacc[i] = 0.f;
        #pragma unroll
        for (int u = 0; u < 2; ++u)
            #pragma unroll
            for (int r = 0; r < 8; ++r) {
                int t = tb + u * 16 + 8 * hf + r;
                Mv[u][r] = gM[t];
                iZ[u][r] = 1.f / gZ[t];
            }
    }

#if HAVE_TDM
    const unsigned long long ehB = (unsigned long long)(size_t)eh;
    const unsigned ldsB = (unsigned)(size_t)(const void*)smem;  // LDS byte addr (low 32 bits)
    if (w == 0) {                       // prime both buffers via the TDM
        tdm_load_tile(ehB,         ldsB);
        tdm_load_tile(ehB + TILEB, ldsB + TILEB);
        __builtin_amdgcn_s_wait_tensorcnt(1);   // buffer 0 complete
    }
#endif
    __syncthreads();

    for (int it = 0; it < NITER; ++it) {
        const int kb = it * KSTEP;
        const int n0 = kb + wk * 16;
        const int k  = n0 + ln;
        const float eq = esq[k];        // issue early, used after the WMMAs

#if HAVE_TDM
        // B row from the current LDS tile (ds_load path, no loadcnt stalls)
        const _Float16* brow = (const _Float16*)(smem + (it & 1) * TILEB)
                             + (wk * 16 + ln) * DIM + 8 * hf;
#else
        // fallback: stream B straight from L2
        const _Float16* brow = eh + (size_t)(n0 + ln) * DIM + 8 * hf;
        if (kb + KSTEP < KC) __builtin_prefetch(brow + KSTEP * DIM, 0, 0);
#endif

        v8f a0 = {}, a1 = {};
        #pragma unroll
        for (int c = 0; c < 8; ++c) {
            v8h blo = *(const v8h*)(brow + 32 * c);
            v8h bhi = *(const v8h*)(brow + 32 * c + 16);
            v16h b = __builtin_shufflevector(blo, bhi, 0,1,2,3,4,5,6,7,8,9,10,11,12,13,14,15);
            a0 = __builtin_amdgcn_wmma_f32_16x16x32_f16(false, af0[c], false, b,
                                                        (short)0, a0, false, false);
            a1 = __builtin_amdgcn_wmma_f32_16x16x32_f16(false, af1[c], false, b,
                                                        (short)0, a1, false, false);
        }

        #pragma unroll
        for (int r = 0; r < 8; ++r) {
            float s0 = fmaf(2.f, a0[r], -eq);       // s = 2 x.e - ||e||^2
            float s1 = fmaf(2.f, a1[r], -eq);
            if (PASS == 1) {
                float d0  = s0 - mx[0][r];
                float ex0 = __expf(-fabsf(d0));
                if (d0 > 0.f) { zs[0][r] = fmaf(zs[0][r], ex0, 1.f); mx[0][r] = s0; bi[0][r] = k; }
                else          { zs[0][r] += ex0; }
                float d1  = s1 - mx[1][r];
                float ex1 = __expf(-fabsf(d1));
                if (d1 > 0.f) { zs[1][r] = fmaf(zs[1][r], ex1, 1.f); mx[1][r] = s1; bi[1][r] = k; }
                else          { zs[1][r] += ex1; }
            } else {
                // both tiles share the same k -> one LDS atomic for two tokens
                float p = __expf(s0 - Mv[0][r]) * iZ[0][r]
                        + __expf(s1 - Mv[1][r]) * iZ[1][r];
                atomicAdd(&lacc[k], p);             // ds_add_f32
            }
        }

#if HAVE_TDM
        __syncthreads();                 // all waves done reading buffer (it&1)
        if (w == 0) {
            if (it + 2 < NITER) {        // refill the just-drained buffer
                tdm_load_tile(ehB + (unsigned long long)(it + 2) * TILEB,
                              ldsB + (unsigned)(it & 1) * TILEB);
                __builtin_amdgcn_s_wait_tensorcnt(1);  // tile for it+1 complete
            } else {
                __builtin_amdgcn_s_wait_tensorcnt(0);
            }
        }
        __syncthreads();                 // publish: next buffer ready
#endif
    }

    if (PASS == 1) {
        // butterfly across the 16 lanes of each half (same token, different k)
        #pragma unroll
        for (int u = 0; u < 2; ++u)
            #pragma unroll
            for (int r = 0; r < 8; ++r) {
                float m = mx[u][r], z = zs[u][r];
                int   b = bi[u][r];
                #pragma unroll
                for (int off = 1; off <= 8; off <<= 1) {
                    float mo = __shfl_xor(m, off, 32);
                    float zo = __shfl_xor(z, off, 32);
                    int   bo = __shfl_xor(b, off, 32);
                    float mn = fmaxf(m, mo);
                    z = z * __expf(m - mn) + zo * __expf(mo - mn);
                    if (mo > m) b = bo;
                    m = mn;
                }
                if (ln == 0) {
                    int tl = wm * 32 + u * 16 + 8 * hf + r;
                    sM[tl][wk] = m; sZ[tl][wk] = z; sB[tl][wk] = b;
                }
            }
        __syncthreads();
        if (tid < 128) {                              // combine the 2 wk partials
            float m = -INFINITY, z = 0.f; int b = 0;
            #pragma unroll
            for (int j = 0; j < 2; ++j) {
                float mo = sM[tid][j], zo = sZ[tid][j];
                float mn = fmaxf(m, mo);
                z = z * __expf(m - mn) + zo * __expf(mo - mn);
                if (mo > m) b = sB[tid][j];
                m = mn;
            }
            int t = blockIdx.x * 128 + tid;
            gM[t] = m; gZ[t] = z; gI[t] = b;
        }
    } else {
        __syncthreads();
        for (int i = tid; i < KC; i += 256) atomicAdd(&avg[i], lacc[i]);
    }
}

// -------------------------------------------------------------------------
// Kernel 4: gather quantized rows, write quantized_st + indices, reduce MSE
// -------------------------------------------------------------------------
__global__ __launch_bounds__(256) void vq_quant(const float* __restrict__ x,
                                                const float* __restrict__ e,
                                                const int* __restrict__ gI,
                                                float* __restrict__ outQ,
                                                float* __restrict__ outIdx,
                                                float* __restrict__ accSq) {
    __shared__ float red[8];
    int i = blockIdx.x * 256 + threadIdx.x;
    int t = i >> 8, d = i & 255;
    int k = gI[t];
    float q = e[(size_t)k * DIM + d];
    outQ[i] = q;                                   // quantized_st == quantized
    if (d == 0) outIdx[t] = (float)k;
    float df = q - x[i];
    float v = df * df;
    #pragma unroll
    for (int off = 16; off >= 1; off >>= 1) v += __shfl_down(v, off, 32);
    int lane = threadIdx.x & 31, wv = threadIdx.x >> 5;
    if (lane == 0) red[wv] = v;
    __syncthreads();
    if (threadIdx.x == 0) {
        float s = 0.f;
        #pragma unroll
        for (int j = 0; j < 8; ++j) s += red[j];
        atomicAdd(accSq, s);
    }
}

// -------------------------------------------------------------------------
// Kernel 5: scalar epilogue — entropy over avg_probs, total loss, perplexity
// -------------------------------------------------------------------------
__global__ __launch_bounds__(256) void vq_final(const float* __restrict__ avg,
                                                const float* __restrict__ accSq,
                                                float* __restrict__ outLoss,
                                                float* __restrict__ outPerp) {
    __shared__ float red[8];
    float ne = 0.f;
    for (int k = threadIdx.x; k < KC; k += 256) {
        float p = avg[k] * (1.f / (float)TOK);
        ne += p * logf(p + 1e-10f);
    }
    #pragma unroll
    for (int off = 16; off >= 1; off >>= 1) ne += __shfl_down(ne, off, 32);
    int lane = threadIdx.x & 31, wv = threadIdx.x >> 5;
    if (lane == 0) red[wv] = ne;
    __syncthreads();
    if (threadIdx.x == 0) {
        float tot = 0.f;
        #pragma unroll
        for (int j = 0; j < 8; ++j) tot += red[j];
        float mse = accSq[0] * (1.f / (float)TD);
        float loss = 1.25f * mse + 0.1f * tot;     // q_loss + 0.25*e_loss + 0.1*negH
        outLoss[0] = loss;
        outPerp[0] = __expf(-tot);
    }
}

// -------------------------------------------------------------------------
extern "C" void kernel_launch(void* const* d_in, const int* in_sizes, int n_in,
                              void* d_out, int out_size, void* d_ws, size_t ws_size,
                              hipStream_t stream) {
    const float* x = (const float*)d_in[0];   // [8,2048,256]
    const float* e = (const float*)d_in[1];   // [8192,256]

    // workspace layout (offsets in bytes)
    char* ws = (char*)d_ws;
    _Float16* xh   = (_Float16*)(ws);                          // 8 MB
    _Float16* eh   = (_Float16*)(ws + 8388608);                // 4 MB
    float*    esq  = (float*)   (ws + 12582912);               // 32 KB
    float*    gM   = (float*)   (ws + 12615680);               // 64 KB
    float*    gZ   = (float*)   (ws + 12681216);               // 64 KB
    int*      gI   = (int*)     (ws + 12746752);               // 64 KB
    float*    avg  = (float*)   (ws + 12812288);               // 32 KB
    float*    aSq  = (float*)   (ws + 12845056);               // 4 B

    // output layout: quantized_st | total_loss | indices | perplexity
    float* o       = (float*)d_out;
    float* outQ    = o;                 // TD floats
    float* outLoss = o + TD;            // 1
    float* outIdx  = o + TD + 1;        // TOK floats
    float* outPerp = o + TD + 1 + TOK;  // 1

    vq_prep<<<TD / 256, 256, 0, stream>>>(x, e, xh, eh, esq, avg, aSq);
    vq_gemm<1><<<TOK / 128, 256, 2 * TILEB,         stream>>>(xh, eh, esq, gM, gZ, gI, avg);
    vq_gemm<2><<<TOK / 128, 256, 2 * TILEB + 32768, stream>>>(xh, eh, esq, gM, gZ, gI, avg);
    vq_quant<<<TD / 256, 256, 0, stream>>>(x, e, gI, outQ, outIdx, aSq);
    vq_final<<<1, 256, 0, stream>>>(avg, aSq, outLoss, outPerp);
}
